// FullAttention_56573309223404
// MI455X (gfx1250) — matrix operand order repaired and checked by reading, not compiled, not run on hardware
//
#include <hip/hip_runtime.h>

typedef __bf16          v16bf __attribute__((ext_vector_type(16)));
typedef float           v8f   __attribute__((ext_vector_type(8)));
typedef unsigned short  us8   __attribute__((ext_vector_type(8)));
typedef unsigned short  us4   __attribute__((ext_vector_type(4)));

// Problem constants (from reference setup_inputs)
constexpr int Bc = 4, Lc = 2048, Hc = 16, Ec = 64;
constexpr int BQ  = 128;     // query rows per block (16 per wave, 8 waves)
constexpr int SCH = 64;      // kv rows per chunk (two K=32 of the PV WMMA)
constexpr long RS = (long)Hc * Ec;   // row stride in floats

__device__ __forceinline__ unsigned short f2bf(float f) {
    __bf16 h = (__bf16)f;                       // native v_cvt_*bf16_f32
    return __builtin_bit_cast(unsigned short, h);
}

union AV {
    us8            h[2];
    unsigned short s[16];
    v16bf          v;
};

// One 64-s flash-attention step for one wave (16 q rows).
template <bool MASK>
__device__ __forceinline__ void attn_step(const unsigned short (*Ksh)[72],
                                          const unsigned short (*Vts)[80],
                                          const v16bf* qb,
                                          int lq, int half, int q, int sbase,
                                          float& m_run, float& l_run, v8f (&o)[4])
{
    const float NEGINF = -__builtin_inff();

    // --- S^T = K * Q^T (scores already in exp2 domain; scale folded in Q) ---
    float st[32];
    #pragma unroll
    for (int sub = 0; sub < 4; ++sub) {
        v8f acc;
        #pragma unroll
        for (int i = 0; i < 8; ++i) acc[i] = 0.0f;
        const int row = sub * 16 + lq;
        #pragma unroll
        for (int w = 0; w < 2; ++w) {
            AV a;   // A-operand: lane holds row, K(e)={b..b+16}
            const int e0 = w * 32 + half * 16;
            a.h[0] = *(const us8*)&Ksh[row][e0];
            a.h[1] = *(const us8*)&Ksh[row][e0 + 8];
            acc = __builtin_amdgcn_wmma_f32_16x16x32_bf16(
                      false, a.v, false, qb[w], (short)0, acc, false, false);
        }
        #pragma unroll
        for (int i = 0; i < 8; ++i) st[sub * 8 + i] = acc[i];
    }

    if (MASK) {  // only the diagonal chunk pays for this
        #pragma unroll
        for (int sub = 0; sub < 4; ++sub)
            #pragma unroll
            for (int i = 0; i < 8; ++i) {
                const int sg = sbase + sub * 16 + i + 8 * half;  // C-layout row
                if (sg > q) st[sub * 8 + i] = NEGINF;
            }
    }

    // --- online softmax (tree reductions) -----------------------------------
    float mx[16];
    #pragma unroll
    for (int k = 0; k < 16; ++k) mx[k] = fmaxf(st[k], st[k + 16]);
    #pragma unroll
    for (int s2 = 8; s2 > 0; s2 >>= 1)
        #pragma unroll
        for (int k = 0; k < s2; ++k) mx[k] = fmaxf(mx[k], mx[k + s2]);
    float tmax = fmaxf(mx[0], __shfl_xor(mx[0], 16, 32));    // full row max

    float m_new = m_run;
    if (__ballot(tmax > m_run) != 0ull) {               // wave-uniform rescale
        m_new = fmaxf(m_run, tmax);
        const float alpha = __builtin_amdgcn_exp2f(m_run - m_new);
        m_run = m_new;
        l_run *= alpha;
        float alf[8];
        #pragma unroll
        for (int i = 0; i < 8; ++i) alf[i] = __shfl(alpha, i + 8 * half, 32);
        #pragma unroll
        for (int dt = 0; dt < 4; ++dt)
            #pragma unroll
            for (int i = 0; i < 8; ++i) o[dt][i] *= alf[i];
    }

    AV pa[2];   // S^T C-layout IS the A-operand layout of P: pack directly
    float ps[8];
    #pragma unroll
    for (int k = 0; k < 8; ++k) ps[k] = 0.0f;
    #pragma unroll
    for (int k = 0; k < 32; ++k) {
        const float p = __builtin_amdgcn_exp2f(st[k] - m_new);
        ps[k & 7] += p;
        pa[k >> 4].s[k & 15] = f2bf(p);
    }
    #pragma unroll
    for (int s2 = 4; s2 > 0; s2 >>= 1)
        #pragma unroll
        for (int k = 0; k < s2; ++k) ps[k] += ps[k + s2];
    l_run += ps[0] + __shfl_xor(ps[0], 16, 32);

    // --- O += P * V ---------------------------------------------------------
    #pragma unroll
    for (int dt = 0; dt < 4; ++dt) {
        const int d = dt * 16 + lq;
        #pragma unroll
        for (int w2 = 0; w2 < 2; ++w2) {
            AV bv;  // B-operand: lane n holds column d of V, s={b..b+8, b+16..b+24}
            const int sb = w2 * 32 + half * 8;
            bv.h[0] = *(const us8*)&Vts[d][sb];
            bv.h[1] = *(const us8*)&Vts[d][sb + 16];
            o[dt] = __builtin_amdgcn_wmma_f32_16x16x32_bf16(
                        false, pa[w2].v, false, bv.v, (short)0, o[dt], false, false);
        }
    }
}

__global__ __launch_bounds__(256)
void fa_fwd_kernel(const float* __restrict__ Q, const float* __restrict__ K,
                   const float* __restrict__ V, float* __restrict__ O)
{
    __shared__ unsigned short Ksh[64][72];   // K chunk [s][e], bf16, padded
    __shared__ unsigned short Vts[64][80];   // V chunk transposed [d][s], bf16

    const int tid  = threadIdx.x;
    const int lane = tid & 31;
    const int wave = tid >> 5;
    const int lq   = lane & 15;
    const int half = lane >> 4;

    const int q0b = blockIdx.x * BQ;
    const int h   = blockIdx.y;
    const int b   = blockIdx.z;
    const int q0w = q0b + wave * 16;

    const float* Qb = Q + (((long)b * Lc) * Hc + h) * Ec;
    const float* Kb = K + (((long)b * Lc) * Hc + h) * Ec;
    const float* Vb = V + (((long)b * Lc) * Hc + h) * Ec;
    float*       Ob = O + (((long)b * Lc) * Hc + h) * Ec;

    const float kscale = 0.125f * 1.44269504088896340736f; // 1/sqrt(E)*log2(e)

    // Q tile in registers, pre-scaled, as B-operand of S^T = K * Q^T
    v16bf qb[2];
    {
        const float* qrow = Qb + (long)(q0w + lq) * RS;
        #pragma unroll
        for (int w = 0; w < 2; ++w) {
            AV av;
            const float4* p = (const float4*)(qrow + w * 32 + half * 16);
            #pragma unroll
            for (int j = 0; j < 4; ++j) {
                float4 x = p[j];
                av.s[j*4+0] = f2bf(x.x * kscale); av.s[j*4+1] = f2bf(x.y * kscale);
                av.s[j*4+2] = f2bf(x.z * kscale); av.s[j*4+3] = f2bf(x.w * kscale);
            }
            qb[w] = av.v;
        }
    }

    // Persistent loader pointers (strength-reduced: advance by one chunk/iter)
    const bool isK = (tid < 128);
    int krow = 0, kcb = 0, vs0 = 0, vd0 = 0;
    const float* ldp;
    if (isK) {
        krow = tid & 63;                // LDS row (s)
        kcb  = (tid >> 6) * 16;         // e base; also handles kcb+32
        ldp  = Kb + (long)krow * RS + kcb;
    } else {
        const int t = tid - 128;
        vs0 = (t & 15) * 4;             // s base (lane-fast: conflict-free DS)
        vd0 = ((t >> 4) & 7) * 4;       // d base; also handles vd0+32
        ldp = Vb + (long)vs0 * RS + vd0;
    }

    float m_run = -__builtin_inff();
    float l_run = 0.0f;
    v8f o[4];
    #pragma unroll
    for (int dt = 0; dt < 4; ++dt)
        #pragma unroll
        for (int i = 0; i < 8; ++i) o[dt][i] = 0.0f;

    const int q     = q0w + lq;             // this lane's query row
    const int nC    = (q0b >> 6) + 2;       // 64-row chunks the block stages
    const int nTot  = (q0w + 79) >> 6;      // chunks this wave computes
    const int nFull = (q0w + 1) >> 6;       // of which, unmasked

    for (int c = 0; c < nC; ++c) {
        const int sbase = c * SCH;
        __syncthreads();                    // prior LDS reads complete
        if (isK) {
            #pragma unroll
            for (int g = 0; g < 2; ++g) {   // two 16-col groups per thread
                const float4* kp = (const float4*)(ldp + g * 32);
                unsigned short kt[16];
                #pragma unroll
                for (int j = 0; j < 4; ++j) {
                    float4 x = kp[j];
                    kt[j*4+0] = f2bf(x.x); kt[j*4+1] = f2bf(x.y);
                    kt[j*4+2] = f2bf(x.z); kt[j*4+3] = f2bf(x.w);
                }
                *(us8*)&Ksh[krow][kcb + g*32]     = *(const us8*)&kt[0];
                *(us8*)&Ksh[krow][kcb + g*32 + 8] = *(const us8*)&kt[8];
            }
        } else {
            #pragma unroll
            for (int g = 0; g < 2; ++g) {   // two 4-col groups per thread
                const int dd = vd0 + g * 32;
                float4 r[4];
                #pragma unroll
                for (int i = 0; i < 4; ++i)
                    r[i] = *(const float4*)(ldp + (long)i * RS + g * 32);
                #pragma unroll
                for (int j = 0; j < 4; ++j) {   // transpose in registers
                    us4 p = { f2bf(((const float*)&r[0])[j]),
                              f2bf(((const float*)&r[1])[j]),
                              f2bf(((const float*)&r[2])[j]),
                              f2bf(((const float*)&r[3])[j]) };
                    *(us4*)&Vts[dd + j][vs0] = p;
                }
            }
        }
        __syncthreads();

        if (c + 1 < nC) {                   // prefetch next chunk during compute
            __builtin_prefetch(ldp + SCH * RS, 0, 0);
            __builtin_prefetch(ldp + SCH * RS + 32, 0, 0);
        }

        if (c < nFull)
            attn_step<false>(Ksh, Vts, qb, lq, half, q, sbase, m_run, l_run, o);
        else if (c < nTot)
            attn_step<true >(Ksh, Vts, qb, lq, half, q, sbase, m_run, l_run, o);

        ldp += SCH * RS;                    // advance one chunk
    }

    // --- epilogue: divide by row sum, store ---------------------------------
    const float rinv = __builtin_amdgcn_rcpf(l_run);
    float ri[8];
    #pragma unroll
    for (int i = 0; i < 8; ++i) ri[i] = __shfl(rinv, i + 8 * half, 32);

    #pragma unroll
    for (int dt = 0; dt < 4; ++dt) {
        const int d = dt * 16 + lq;
        #pragma unroll
        for (int i = 0; i < 8; ++i) {
            const int l = i + 8 * half;
            Ob[(long)(q0w + l) * RS + d] = o[dt][i] * ri[i];
        }
    }
}

extern "C" void kernel_launch(void* const* d_in, const int* in_sizes, int n_in,
                              void* d_out, int out_size, void* d_ws, size_t ws_size,
                              hipStream_t stream) {
    (void)in_sizes; (void)n_in; (void)out_size; (void)d_ws; (void)ws_size;
    const float* Q = (const float*)d_in[0];
    const float* K = (const float*)d_in[1];
    const float* V = (const float*)d_in[2];
    float*       O = (float*)d_out;

    dim3 grid(Lc / BQ, Hc, Bc);   // (16, 16, 4)
    dim3 block(256);              // 8 wave32s
    fa_fwd_kernel<<<grid, block, 0, stream>>>(Q, K, V, O);
}
